// VAE_conv_ENCODER_19464791786169
// MI455X (gfx1250) — matrix operand order repaired and checked by reading, not compiled, not run on hardware
//
#include <hip/hip_runtime.h>
#include <math.h>

// Problem constants (match reference)
#define NN      20000     // nodes (divisible by 32 -> clean 32-row M stripes)
#define EE      320000    // edges
#define BBATCH  256       // graphs in batch
#define DD      64        // feature dim
#define HH      4         // heads
#define HCC     256      // H*D
#define QW      832       // fused GEMM width: 3*HC (q,k,v) + D (skip)
#define KDIM    64        // GEMM K
#define NTILES  52        // QW/16
#define TNHALF  26        // n-tiles per half panel
#define FRAG_ELEMS 512    // elems per (tile,k-step) fragment: 32 lanes x 16
#define HALF_ELEMS (TNHALF * 2 * FRAG_ELEMS)   // 26,624 bf16 = 53,248 B
#define LAYER_ELEMS (NTILES * 2 * FRAG_ELEMS)  // 53,248 bf16 per layer
#define ATT_SCALE 0.125f  // 1/sqrt(64)

typedef __attribute__((ext_vector_type(16))) __bf16 v16bf;
typedef __attribute__((ext_vector_type(8)))  float  v8f;
typedef __attribute__((ext_vector_type(4)))  int    v4i;
typedef __attribute__((ext_vector_type(8)))  int    v8i;
typedef __attribute__((address_space(1))) v4i* gptr_b128;   // global int4*
typedef __attribute__((address_space(3))) v4i* lptr_b128;   // LDS int4*

// feature detection for gfx1250 async global->LDS path
#if defined(__has_builtin)
#if __has_builtin(__builtin_amdgcn_global_load_async_to_lds_b128) && \
    __has_builtin(__builtin_amdgcn_s_wait_asynccnt)
#define USE_ASYNC_LDS 1
#endif
#if __has_builtin(__builtin_amdgcn_s_cluster_barrier)
#define HAVE_CLUSTER_BARRIER 1
#endif
#endif
#ifndef USE_ASYNC_LDS
#define USE_ASYNC_LDS 0
#endif
#ifndef HAVE_CLUSTER_BARRIER
#define HAVE_CLUSTER_BARRIER 0
#endif

// ---------- helpers ----------
__device__ __forceinline__ unsigned fenc(float f) {
  unsigned u = __float_as_uint(f);
  return (u & 0x80000000u) ? ~u : (u | 0x80000000u);
}
__device__ __forceinline__ float fdec(unsigned e) {
  unsigned u = (e & 0x80000000u) ? (e & 0x7fffffffu) : ~e;
  return __uint_as_float(u);
}
__device__ __forceinline__ float sigm(float x) { return 1.0f / (1.0f + __expf(-x)); }
__device__ __forceinline__ unsigned pcg(unsigned v) {
  v = v * 747796405u + 2891336453u;
  unsigned w = ((v >> ((v >> 28u) + 4u)) ^ v) * 277803737u;
  return (w >> 22u) ^ w;
}
__device__ __forceinline__ unsigned lds_off(void* p) {
  // raw LDS byte offset of a __shared__ object (AS3 ptrtoint)
  return (unsigned)(unsigned long long)(__attribute__((address_space(3))) void*)p;
}

// ---------- utility kernels ----------
__global__ void k_fill_f32(float* p, float v, int n) {
  int i = blockIdx.x * blockDim.x + threadIdx.x;
  if (i < n) p[i] = v;
}
__global__ void k_fill_u32(unsigned* p, unsigned v, int n) {
  int i = blockIdx.x * blockDim.x + threadIdx.x;
  if (i < n) p[i] = v;
}
__global__ void k_copy_f32(float* dst, const float* src, int n) {
  int i = blockIdx.x * blockDim.x + threadIdx.x;
  if (i < n) dst[i] = src[i];
}

// ---------- weight fusion + bf16 conversion, packed in WMMA fragment order ----
// Bfrag layout: [layer][nt 0..51][s 0..1][lane 0..31][slot 0..15]
//   K  = s*32 + slot + 16*(lane>>4)     (contiguous k order, same as A)
//   col= nt*16 + (lane&15);  col maps: 0..255 Wq | 256..511 Wk | 512..767 Wv | 768..831 Wskip
__global__ void k_convert_weights(const float* __restrict__ Wq, const float* __restrict__ bq,
                                  const float* __restrict__ Wk, const float* __restrict__ bk,
                                  const float* __restrict__ Wv, const float* __restrict__ bv,
                                  const float* __restrict__ Wskip, const float* __restrict__ bskip,
                                  __bf16* __restrict__ Bfrag, float* __restrict__ biascat) {
  int idx = blockIdx.x * blockDim.x + threadIdx.x;
  const int total = 4 * LAYER_ELEMS;
  if (idx >= total) return;
  int slot = idx & 15;
  int lane = (idx >> 4) & 31;
  int s    = (idx >> 9) & 1;
  int nt   = (idx >> 10) % NTILES;
  int l    = (idx >> 10) / NTILES;
  int k = s * 32 + slot + 16 * (lane >> 4);
  int c = nt * 16 + (lane & 15);
  float wv, bvv;
  if (c < 256)      { wv = Wq[(l * KDIM + k) * HCC + c];           bvv = bq[l * HCC + c]; }
  else if (c < 512) { wv = Wk[(l * KDIM + k) * HCC + (c - 256)];   bvv = bk[l * HCC + (c - 256)]; }
  else if (c < 768) { wv = Wv[(l * KDIM + k) * HCC + (c - 512)];   bvv = bv[l * HCC + (c - 512)]; }
  else              { wv = Wskip[(l * KDIM + k) * DD + (c - 768)]; bvv = bskip[l * DD + (c - 768)]; }
  Bfrag[idx] = (__bf16)wv;
  if (k == 0) biascat[l * QW + c] = bvv;
}

// ---------- WMMA GEMM: TDM A stripe + async fragment-major B panel ----------
// Grid: (NN/32, 2). Block = 256 threads = 8 waves.
//  * wave 0 issues a 1-D TENSOR_LOAD_TO_LDS for the 32x64 f32 A stripe (8 KB)
//  * all waves async-copy the 53 KB fragment-major B half-panel to LDS
//  * each wave computes ~6-7 16x16 tiles; B fragments are single contiguous
//    32 B LDS reads; A fragments come from the TDM-staged LDS stripe.
__global__ __launch_bounds__(256) void k_gemm_wmma(
    const float* __restrict__ A, const __bf16* __restrict__ Bfrag,
    const float* __restrict__ bias, float* __restrict__ C) {
  __shared__ __attribute__((aligned(64))) __bf16 ldsB[HALF_ELEMS];  // 53,248 B
  __shared__ __attribute__((aligned(64))) float  ldsA[32 * KDIM];   // 8,192 B
  const int tid = threadIdx.x;
  const int mstripe = blockIdx.x;          // 32 rows
  const int halfsel = blockIdx.y;          // which half of the 832 columns
  const __bf16* Bsrc = Bfrag + (size_t)halfsel * HALF_ELEMS;

  __builtin_prefetch(Bsrc, 0, 3);          // global_prefetch_b8

  // ---- TDM: DMA the contiguous 8 KB A stripe into LDS (wave 0 only) ----
  if (tid < 32) {
    unsigned long long ga = (unsigned long long)(A + (size_t)mstripe * 32 * KDIM);
    unsigned lofs = lds_off(ldsA);
    // D# group0: count=1 | lds_addr | global_addr[56:0] | type=2
    v4i g0 = { 1, (int)lofs, (int)(unsigned)(ga & 0xffffffffu),
               (int)((unsigned)((ga >> 32) & 0x01ffffffu) | 0x80000000u) };
    // D# group1: 1-D tile: data_size=4B(code 2), tensor_dim0=tile_dim0=2048
    v8i g1 = { (int)(2u << 16),          // workgroup_mask=0, data_size=2
               (int)(2048u << 16),       // tensor_dim0[15:0] @ bits 63:48
               (int)(1u << 16),          // tensor_dim0 hi=0, tensor_dim1=1
               (int)(2048u << 16),       // tile_dim0 @ bits 127:112
               0,                        // tile_dim1=0 (unused), tile_dim2=0
               2048,                     // tensor_dim0_stride lo
               0, 0 };
    asm volatile("tensor_load_to_lds %0, %1\n\t"
                 "s_wait_tensorcnt 0x0"
                 :: "s"(g0), "s"(g1) : "memory");
  }

  // ---- async-copy B half-panel (linear, fragment-major) ----
  {
    const int chunks = HALF_ELEMS / 8;   // 3328 x 16 B
    for (int i = tid; i < chunks; i += 256) {
      const __bf16* g = Bsrc + i * 8;
      __bf16* l = &ldsB[i * 8];
#if USE_ASYNC_LDS
      __builtin_amdgcn_global_load_async_to_lds_b128(
          (gptr_b128)(void*)g, (lptr_b128)(void*)l, 0, 0);
#else
      *(uint4*)l = *(const uint4*)g;
#endif
    }
#if USE_ASYNC_LDS
    __builtin_amdgcn_s_wait_asynccnt(0);
#endif
    __syncthreads();
#if HAVE_CLUSTER_BARRIER
    __builtin_amdgcn_s_cluster_barrier();  // NOP when not dispatched in clusters
#endif
  }

  const int w = tid >> 5, lane = tid & 31;
  const int hlf = lane >> 4, l15 = lane & 15;

  int mt_cached = -1;
  v16bf a0, a1;
  for (int t = w; t < 2 * TNHALF; t += 8) {      // wave-uniform loop
    int mt = t / TNHALF;                         // 0 or 1 (wave-uniform)
    int nt = t - mt * TNHALF;
    if (mt != mt_cached) {                       // wave-uniform branch
      mt_cached = mt;
      int rowL = mt * 16 + l15;                  // row within the LDS stripe
#pragma unroll
      for (int i = 0; i < 16; ++i) {
        // A k order used here (contiguous, same as B): half0 slots 0-15 -> K 0-15;
        //                      half1 slots 0-15 -> K 16-31.
        int ka = i + 16 * hlf;
        a0[i] = (__bf16)ldsA[rowL * KDIM + ka];
        a1[i] = (__bf16)ldsA[rowL * KDIM + 32 + ka];
      }
    }
    // fragment-major B: one contiguous 32 B read per k-step
    v16bf b0 = *(const v16bf*)&ldsB[((nt * 2 + 0) * 32 + lane) * 16];
    v16bf b1 = *(const v16bf*)&ldsB[((nt * 2 + 1) * 32 + lane) * 16];
    v8f acc = {};
    acc = __builtin_amdgcn_wmma_f32_16x16x32_bf16(false, a0, false, b0,
                                                  (short)0, acc, false, false);
    acc = __builtin_amdgcn_wmma_f32_16x16x32_bf16(false, a1, false, b1,
                                                  (short)0, acc, false, false);
    int col = halfsel * 416 + nt * 16 + l15;
    float bc = bias[col];
#pragma unroll
    for (int r = 0; r < 8; ++r) {
      // C layout: VGPR r -> M = r (lanes 0-15) / 8+r (lanes 16-31)
      C[(size_t)(mstripe * 32 + mt * 16 + r + 8 * hlf) * QW + col] = acc[r] + bc;
    }
  }
}

// ---------- edge attention ----------
__global__ void k_edge_alpha(const float* __restrict__ QKVS, const int* __restrict__ eidx,
                             const float* __restrict__ eattr, const float* __restrict__ We,
                             float* __restrict__ alpha, unsigned* __restrict__ mbuf) {
  int idx = blockIdx.x * blockDim.x + threadIdx.x;
  if (idx >= EE * HH) return;
  int e = idx >> 2, hd = idx & 3;
  int src = eidx[e], dst = eidx[EE + e];
  float ea0 = eattr[e * 4 + 0], ea1 = eattr[e * 4 + 1];
  float ea2 = eattr[e * 4 + 2], ea3 = eattr[e * 4 + 3];
  const float4* q4 = (const float4*)(QKVS + (size_t)dst * QW + hd * DD);
  const float4* k4 = (const float4*)(QKVS + (size_t)src * QW + 256 + hd * DD);
  const float* W0 = We + hd * DD;  // per-layer We is [4, 256]
  float s = 0.f;
#pragma unroll 4
  for (int d4 = 0; d4 < DD / 4; ++d4) {
    float4 q = q4[d4], kk = k4[d4];
    float4 w0 = *(const float4*)&W0[d4 * 4];
    float4 w1 = *(const float4*)&W0[256 + d4 * 4];
    float4 w2 = *(const float4*)&W0[512 + d4 * 4];
    float4 w3 = *(const float4*)&W0[768 + d4 * 4];
    s += q.x * (kk.x + ea0 * w0.x + ea1 * w1.x + ea2 * w2.x + ea3 * w3.x);
    s += q.y * (kk.y + ea0 * w0.y + ea1 * w1.y + ea2 * w2.y + ea3 * w3.y);
    s += q.z * (kk.z + ea0 * w0.z + ea1 * w1.z + ea2 * w2.z + ea3 * w3.z);
    s += q.w * (kk.w + ea0 * w0.w + ea1 * w1.w + ea2 * w2.w + ea3 * w3.w);
  }
  s *= ATT_SCALE;
  alpha[idx] = s;
  atomicMax(&mbuf[dst * HH + hd], fenc(s));
}

__global__ void k_edge_exp(const int* __restrict__ eidx, float* __restrict__ alpha,
                           const unsigned* __restrict__ mbuf, float* __restrict__ den) {
  int idx = blockIdx.x * blockDim.x + threadIdx.x;
  if (idx >= EE * HH) return;
  int e = idx >> 2, hd = idx & 3;
  int dst = eidx[EE + e];
  float ex = __expf(alpha[idx] - fdec(mbuf[dst * HH + hd]));
  alpha[idx] = ex;
  atomicAdd(&den[dst * HH + hd], ex);
}

// one thread per (edge, 4-dim group): float4 loads, 4 atomic adds
__global__ void k_edge_msg(const float* __restrict__ QKVS, const int* __restrict__ eidx,
                           const float* __restrict__ eattr, const float* __restrict__ We,
                           const float* __restrict__ alpha, const float* __restrict__ den,
                           float* __restrict__ outb) {
  int idx = blockIdx.x * blockDim.x + threadIdx.x;
  if (idx >= EE * (DD / 4)) return;
  int e = idx >> 4, d4 = idx & 15;
  int d = d4 * 4;
  int src = eidx[e], dst = eidx[EE + e];
  float ea0 = eattr[e * 4 + 0], ea1 = eattr[e * 4 + 1];
  float ea2 = eattr[e * 4 + 2], ea3 = eattr[e * 4 + 3];
  float ax = 0.f, ay = 0.f, az = 0.f, aw = 0.f;
#pragma unroll
  for (int hd = 0; hd < HH; ++hd) {
    float a = alpha[e * HH + hd] / (den[dst * HH + hd] + 1e-16f);
    const float* W0 = We + hd * DD;
    float4 v  = *(const float4*)&QKVS[(size_t)src * QW + 512 + hd * DD + d];
    float4 w0 = *(const float4*)&W0[d];
    float4 w1 = *(const float4*)&W0[256 + d];
    float4 w2 = *(const float4*)&W0[512 + d];
    float4 w3 = *(const float4*)&W0[768 + d];
    ax += (v.x + ea0 * w0.x + ea1 * w1.x + ea2 * w2.x + ea3 * w3.x) * a;
    ay += (v.y + ea0 * w0.y + ea1 * w1.y + ea2 * w2.y + ea3 * w3.y) * a;
    az += (v.z + ea0 * w0.z + ea1 * w1.z + ea2 * w2.z + ea3 * w3.z) * a;
    aw += (v.w + ea0 * w0.w + ea1 * w1.w + ea2 * w2.w + ea3 * w3.w) * a;
  }
  float* ob = outb + (size_t)dst * DD + d;
  atomicAdd(ob + 0, 0.25f * ax);
  atomicAdd(ob + 1, 0.25f * ay);
  atomicAdd(ob + 2, 0.25f * az);
  atomicAdd(ob + 3, 0.25f * aw);
}

// ---------- beta-gated skip combine + ReLU ----------
__global__ void k_combine(const float* __restrict__ outb, const float* __restrict__ QKVS,
                          const float* __restrict__ Wbeta, float* __restrict__ hnext) {
  int n = blockIdx.x * blockDim.x + threadIdx.x;
  if (n >= NN) return;
  const float* o  = outb + (size_t)n * DD;
  const float* xr = QKVS + (size_t)n * QW + 768;
  float t = 0.f;
#pragma unroll 8
  for (int d = 0; d < DD; ++d) {
    float od = o[d], xd = xr[d];
    t += od * Wbeta[d] + xd * Wbeta[DD + d] + (od - xd) * Wbeta[2 * DD + d];
  }
  float beta = sigm(t);
#pragma unroll 8
  for (int d = 0; d < DD; ++d) {
    float v = beta * xr[d] + (1.f - beta) * o[d];
    hnext[(size_t)n * DD + d] = fmaxf(v, 0.f);
  }
}

// ---------- batch norm ----------
__global__ void k_bn_stats(const float* __restrict__ hb, float* __restrict__ stats) {
  int c = blockIdx.x, tid = threadIdx.x;
  __shared__ float s1[256], s2[256];
  float a = 0.f, b = 0.f;
  for (int n = tid; n < NN; n += 256) {
    float v = hb[(size_t)n * DD + c];
    a += v; b += v * v;
  }
  s1[tid] = a; s2[tid] = b;
  __syncthreads();
  for (int off = 128; off > 0; off >>= 1) {
    if (tid < off) { s1[tid] += s1[tid + off]; s2[tid] += s2[tid + off]; }
    __syncthreads();
  }
  if (tid == 0) {
    float mu = s1[0] / (float)NN;
    stats[c] = mu;
    stats[DD + c] = s2[0] / (float)NN - mu * mu;
  }
}
__global__ void k_bn_apply(float* __restrict__ hb, const float* __restrict__ stats,
                           const float* __restrict__ gamma, const float* __restrict__ beta) {
  int idx = blockIdx.x * blockDim.x + threadIdx.x;
  if (idx >= NN * DD) return;
  int c = idx & 63;
  hb[idx] = gamma[c] * (hb[idx] - stats[c]) * rsqrtf(stats[DD + c] + 1e-5f) + beta[c];
}

// ---------- set2set ----------
__global__ void k_gates(const float* __restrict__ qstar, const float* __restrict__ hS,
                        const float* __restrict__ Wih, const float* __restrict__ Whh,
                        const float* __restrict__ bih, const float* __restrict__ bhh,
                        float* __restrict__ gates) {
  int idx = blockIdx.x * blockDim.x + threadIdx.x;
  if (idx >= BBATCH * 256) return;
  int b = idx >> 8, g = idx & 255;
  float acc = bih[g] + bhh[g];
  const float* q  = qstar + b * 128;
  const float* Wg = Wih + g * 128;
  for (int k = 0; k < 128; ++k) acc += q[k] * Wg[k];
  const float* hp = hS + b * 64;
  const float* Wh = Whh + g * 64;
  for (int k = 0; k < 64; ++k) acc += hp[k] * Wh[k];
  gates[idx] = acc;
}
__global__ void k_lstm(const float* __restrict__ gates, float* __restrict__ hS,
                       float* __restrict__ cS) {
  int idx = blockIdx.x * blockDim.x + threadIdx.x;
  if (idx >= BBATCH * DD) return;
  int b = idx >> 6, d = idx & 63;
  const float* g = gates + b * 256;
  float ii = sigm(g[d]), ff = sigm(g[64 + d]);
  float gg = tanhf(g[128 + d]), oo = sigm(g[192 + d]);
  float c = ff * cS[idx] + ii * gg;
  cS[idx] = c;
  hS[idx] = oo * tanhf(c);
}
__global__ void k_attn_e(const float* __restrict__ hb, const float* __restrict__ hS,
                         const int* __restrict__ batch, float* __restrict__ exN,
                         unsigned* __restrict__ mbufB) {
  int n = blockIdx.x * blockDim.x + threadIdx.x;
  if (n >= NN) return;
  int b = batch[n];
  const float4* x4 = (const float4*)(hb + (size_t)n * DD);
  const float4* h4 = (const float4*)(hS + (size_t)b * DD);
  float s = 0.f;
#pragma unroll
  for (int d4 = 0; d4 < DD / 4; ++d4) {
    float4 x = x4[d4], h = h4[d4];
    s += x.x * h.x + x.y * h.y + x.z * h.z + x.w * h.w;
  }
  exN[n] = s;
  atomicMax(&mbufB[b], fenc(s));
}
__global__ void k_attn_exp(const int* __restrict__ batch, float* __restrict__ exN,
                           const unsigned* __restrict__ mbufB, float* __restrict__ denB) {
  int n = blockIdx.x * blockDim.x + threadIdx.x;
  if (n >= NN) return;
  int b = batch[n];
  float ex = __expf(exN[n] - fdec(mbufB[b]));
  exN[n] = ex;
  atomicAdd(&denB[b], ex);
}
__global__ void k_attn_r(const float* __restrict__ hb, const int* __restrict__ batch,
                         const float* __restrict__ exN, const float* __restrict__ denB,
                         float* __restrict__ rbuf) {
  int idx = blockIdx.x * blockDim.x + threadIdx.x;
  if (idx >= NN * DD) return;
  int n = idx >> 6, d = idx & 63;
  int b = batch[n];
  float a = exN[n] / (denB[b] + 1e-16f);
  atomicAdd(&rbuf[b * DD + d], a * hb[idx]);
}
__global__ void k_qstar(const float* __restrict__ hS, const float* __restrict__ rbuf,
                        float* __restrict__ qstar) {
  int idx = blockIdx.x * blockDim.x + threadIdx.x;
  if (idx >= BBATCH * 128) return;
  int b = idx >> 7, j = idx & 127;
  qstar[idx] = (j < 64) ? hS[b * 64 + j] : rbuf[b * 64 + (j - 64)];
}

// ---------- final heads + reparameterization ----------
__global__ void k_final(const float* __restrict__ qstar, const float* __restrict__ Wmu,
                        const float* __restrict__ bmu, const float* __restrict__ Wlv,
                        const float* __restrict__ blv, float* __restrict__ out) {
  int idx = blockIdx.x * blockDim.x + threadIdx.x;
  if (idx >= BBATCH * 128) return;
  int b = idx >> 7, j = idx & 127;
  const float* q = qstar + b * 128;
  float mu = bmu[j], lv = blv[j];
  for (int k = 0; k < 128; ++k) {
    float qk = q[k];
    mu += qk * Wmu[k * 128 + j];
    lv += qk * Wlv[k * 128 + j];
  }
  unsigned r1 = pcg((unsigned)idx * 2u + 12345u);
  unsigned r2 = pcg((unsigned)idx * 2u + 1u + 6789u);
  float u1 = ((float)r1 + 1.0f) * 2.3283064365386963e-10f;
  float u2 = (float)r2 * 2.3283064365386963e-10f;
  float eps = sqrtf(-2.f * __logf(u1)) * __cosf(6.28318530718f * u2);
  out[idx]         = eps * __expf(0.5f * lv) + mu;  // z
  out[32768 + idx] = mu;                            // z_mu
  out[65536 + idx] = lv;                            // z_lsgms
}

// ---------- host orchestration ----------
extern "C" void kernel_launch(void* const* d_in, const int* in_sizes, int n_in,
                              void* d_out, int out_size, void* d_ws, size_t ws_size,
                              hipStream_t stream) {
  const float* x     = (const float*)d_in[0];
  const float* eattr = (const float*)d_in[1];
  const int*   eidx  = (const int*)d_in[2];
  const int*   batch = (const int*)d_in[3];
  const float* Wq    = (const float*)d_in[4];
  const float* bq    = (const float*)d_in[5];
  const float* Wk    = (const float*)d_in[6];
  const float* bk    = (const float*)d_in[7];
  const float* Wv    = (const float*)d_in[8];
  const float* bv    = (const float*)d_in[9];
  const float* We    = (const float*)d_in[10];
  const float* Wskip = (const float*)d_in[11];
  const float* bskip = (const float*)d_in[12];
  const float* Wbeta = (const float*)d_in[13];
  const float* bn_g  = (const float*)d_in[14];
  const float* bn_b  = (const float*)d_in[15];
  const float* Wih   = (const float*)d_in[16];
  const float* Whh   = (const float*)d_in[17];
  const float* bih   = (const float*)d_in[18];
  const float* bhh   = (const float*)d_in[19];
  const float* Wmu   = (const float*)d_in[20];
  const float* bmu   = (const float*)d_in[21];
  const float* Wlv   = (const float*)d_in[22];
  const float* blv   = (const float*)d_in[23];
  float* out = (float*)d_out;
  (void)in_sizes; (void)n_in; (void)out_size; (void)ws_size;

  uintptr_t w = (uintptr_t)d_ws;
  auto carve = [&](size_t bytes) -> void* {
    void* p = (void*)w;
    w += (bytes + 255) & ~(size_t)255;
    return p;
  };
  __bf16*   Bfrag   = (__bf16*)carve((size_t)4 * LAYER_ELEMS * sizeof(__bf16));
  float*    biascat = (float*)carve((size_t)4 * QW * sizeof(float));
  float*    h0      = (float*)carve((size_t)NN * DD * 4);
  float*    h1      = (float*)carve((size_t)NN * DD * 4);
  float*    QKVS    = (float*)carve((size_t)NN * QW * 4);
  float*    alphaB  = (float*)carve((size_t)EE * HH * 4);
  unsigned* mbuf    = (unsigned*)carve((size_t)NN * HH * 4);
  float*    den     = (float*)carve((size_t)NN * HH * 4);
  float*    outb    = (float*)carve((size_t)NN * DD * 4);
  float*    stats   = (float*)carve((size_t)2 * DD * 4);
  float*    hS      = (float*)carve((size_t)BBATCH * DD * 4);
  float*    cS      = (float*)carve((size_t)BBATCH * DD * 4);
  float*    qstar   = (float*)carve((size_t)BBATCH * 128 * 4);
  float*    gates   = (float*)carve((size_t)BBATCH * 256 * 4);
  unsigned* mbufB   = (unsigned*)carve((size_t)BBATCH * 4);
  float*    denB    = (float*)carve((size_t)BBATCH * 4);
  float*    exN     = (float*)carve((size_t)NN * 4);
  float*    rbuf    = (float*)carve((size_t)BBATCH * DD * 4);

  const unsigned NEGINF_ENC = 0x007FFFFFu;  // fenc(-inf)
  dim3 blk(256);
  auto g1 = [](int n) { return dim3((unsigned)((n + 255) / 256)); };

  k_convert_weights<<<g1(4 * LAYER_ELEMS), blk, 0, stream>>>(Wq, bq, Wk, bk, Wv, bv,
                                                             Wskip, bskip, Bfrag, biascat);
  k_copy_f32<<<g1(NN * DD), blk, 0, stream>>>(h0, x, NN * DD);

  float* hcur = h0;
  float* hnxt = h1;
  for (int l = 0; l < 4; ++l) {
    k_gemm_wmma<<<dim3(NN / 32, 2), blk, 0, stream>>>(
        hcur, Bfrag + (size_t)l * LAYER_ELEMS, biascat + l * QW, QKVS);
    k_fill_u32<<<g1(NN * HH), blk, 0, stream>>>(mbuf, NEGINF_ENC, NN * HH);
    k_fill_f32<<<g1(NN * HH), blk, 0, stream>>>(den, 0.f, NN * HH);
    k_fill_f32<<<g1(NN * DD), blk, 0, stream>>>(outb, 0.f, NN * DD);
    const float* WeL = We + (size_t)l * 4 * HCC;
    k_edge_alpha<<<g1(EE * HH), blk, 0, stream>>>(QKVS, eidx, eattr, WeL, alphaB, mbuf);
    k_edge_exp<<<g1(EE * HH), blk, 0, stream>>>(eidx, alphaB, mbuf, den);
    k_edge_msg<<<g1(EE * (DD / 4)), blk, 0, stream>>>(QKVS, eidx, eattr, WeL, alphaB, den, outb);
    k_combine<<<g1(NN), blk, 0, stream>>>(outb, QKVS, Wbeta + l * 192, hnxt);
    if (l < 3) {
      k_bn_stats<<<dim3(DD), blk, 0, stream>>>(hnxt, stats);
      k_bn_apply<<<g1(NN * DD), blk, 0, stream>>>(hnxt, stats, bn_g + l * DD, bn_b + l * DD);
    }
    float* t = hcur; hcur = hnxt; hnxt = t;
  }

  // set2set pooling
  k_fill_f32<<<g1(BBATCH * DD), blk, 0, stream>>>(hS, 0.f, BBATCH * DD);
  k_fill_f32<<<g1(BBATCH * DD), blk, 0, stream>>>(cS, 0.f, BBATCH * DD);
  k_fill_f32<<<g1(BBATCH * 128), blk, 0, stream>>>(qstar, 0.f, BBATCH * 128);
  for (int step = 0; step < 4; ++step) {
    k_gates<<<g1(BBATCH * 256), blk, 0, stream>>>(qstar, hS, Wih, Whh, bih, bhh, gates);
    k_lstm<<<g1(BBATCH * DD), blk, 0, stream>>>(gates, hS, cS);
    k_fill_u32<<<g1(BBATCH), blk, 0, stream>>>(mbufB, NEGINF_ENC, BBATCH);
    k_fill_f32<<<g1(BBATCH), blk, 0, stream>>>(denB, 0.f, BBATCH);
    k_fill_f32<<<g1(BBATCH * DD), blk, 0, stream>>>(rbuf, 0.f, BBATCH * DD);
    k_attn_e<<<g1(NN), blk, 0, stream>>>(hcur, hS, batch, exN, mbufB);
    k_attn_exp<<<g1(NN), blk, 0, stream>>>(batch, exN, mbufB, denB);
    k_attn_r<<<g1(NN * DD), blk, 0, stream>>>(hcur, batch, exN, denB, rbuf);
    k_qstar<<<g1(BBATCH * 128), blk, 0, stream>>>(hS, rbuf, qstar);
  }

  k_final<<<g1(BBATCH * 128), blk, 0, stream>>>(qstar, Wmu, bmu, Wlv, blv, out);
}
